// ESMGraphEncoder_16887811407953
// MI455X (gfx1250) — matrix operand order repaired and checked by reading, not compiled, not run on hardware
//
#include <hip/hip_runtime.h>
#include <stdint.h>

#define G_NUM   2048
#define D_DIM   1280
#define P_DIM   128
#define KSTEPS  (D_DIM / 32)   // 40
#define TILE_M  128
#define LN_EPSF 1e-5f

typedef __attribute__((ext_vector_type(16))) __bf16 v16bf;
typedef __attribute__((ext_vector_type(8)))  float  v8f;

// ---- helpers -------------------------------------------------------------

// fp32 -> bf16 (round-to-nearest-even), manual bit version (packW only)
__device__ __forceinline__ unsigned bf16rne(float f) {
  unsigned u = __float_as_uint(f);
  unsigned r = u + 0x7FFFu + ((u >> 16) & 1u);
  return r >> 16;
}
__device__ __forceinline__ unsigned packbf(float lo, float hi) {
  return bf16rne(lo) | (bf16rne(hi) << 16);
}
// order-preserving float<->uint (for atomicMax on floats incl. negatives)
__device__ __forceinline__ unsigned f2ord(float x) {
  unsigned u = __float_as_uint(x);
  return (u & 0x80000000u) ? ~u : (u | 0x80000000u);
}
__device__ __forceinline__ float ord2f(unsigned u) {
  return (u & 0x80000000u) ? __uint_as_float(u & 0x7FFFFFFFu)
                           : __uint_as_float(~u);
}

// CDNA5 async global->LDS copy, 16B per lane (GLOBAL_LOAD_ASYNC_TO_LDS_B128,
// GVS mode: saddr base + 32-bit per-lane offset). Tracked by ASYNCcnt.
__device__ __forceinline__ void async_load_b128(unsigned lds_addr, unsigned voff,
                                                const void* sbase) {
  asm volatile("global_load_async_to_lds_b128 %0, %1, %2"
               :: "v"(lds_addr), "v"(voff), "s"(sbase)
               : "memory");
}
__device__ __forceinline__ void wait_asynccnt0() {
  asm volatile("s_wait_asynccnt 0x0" ::: "memory");
}

// ---- kernel 0: zero the pooling accumulators -----------------------------

__global__ void esm_init_kernel(float* __restrict__ z_sum,
                                float* __restrict__ z_cnt,
                                unsigned* __restrict__ z_maxu) {
  int i = blockIdx.x * 256 + threadIdx.x;
  if (i < G_NUM * P_DIM) { z_sum[i] = 0.f; z_maxu[i] = 0u; }
  if (i < G_NUM)         { z_cnt[i] = 0.f; }
}

// ---- kernel 1: repack W (fp32 [1280][128]) into bf16 B-fragment order ----
// Wp[ks][nt][lane][j] (uint) = pack(W[k][col], W[k+1][col])
//   k = ks*32 + 16*(lane>>4) + 2*j,  col = nt*16 + (lane&15)

__global__ void esm_packW_kernel(const float* __restrict__ W,
                                 unsigned* __restrict__ Wp) {
  int u = blockIdx.x * 256 + threadIdx.x;     // 0 .. 40*2048-1
  int ks   = u >> 11;
  int rem  = u & 2047;
  int nt   = rem >> 8;
  int rem2 = rem & 255;
  int l    = rem2 >> 3;
  int j    = rem2 & 7;
  int k   = ks * 32 + ((l >> 4) << 4) + 2 * j;
  int col = nt * 16 + (l & 15);
  Wp[u] = packbf(W[(size_t)k * P_DIM + col], W[(size_t)(k + 1) * P_DIM + col]);
}

// ---- kernel 2: fused GEMM(bf16 WMMA) + bias + LN + SiLU + segment pool ---

__global__ __launch_bounds__(256)
void esm_gemm_ln_silu_pool_kernel(const float* __restrict__ x,
                                  const int* __restrict__ batch,
                                  const unsigned* __restrict__ Wp,
                                  const float* __restrict__ bias,
                                  const float* __restrict__ gamma,
                                  const float* __restrict__ beta,
                                  float* __restrict__ z_sum,
                                  float* __restrict__ z_cnt,
                                  unsigned* __restrict__ z_maxu,
                                  int N) {
  __shared__ __attribute__((aligned(32))) unsigned Bs[2][2048]; // 2 x 8 KB
  __shared__ float hs[TILE_M][132];                             // 67.6 KB
  __shared__ int gs[TILE_M];

  const int tid  = threadIdx.x;
  const int wave = tid >> 5;
  const int lane = tid & 31;
  const int hlf  = lane >> 4;      // 0/1: K-interleave group
  const int l15  = lane & 15;

  const int rowBase = blockIdx.x * TILE_M;
  int rowA = rowBase + wave * 16 + l15;     // A-matrix row this lane feeds
  if (rowA >= N) rowA = N - 1;              // clamp (keeps EXEC uniform)
  const float* xrow = x + (size_t)rowA * D_DIM;

  // LDS byte address of Bs (flat-LDS aperture: low 32 bits = LDS offset)
  const unsigned bsBase = (unsigned)(uintptr_t)(&Bs[0][0]);

  v8f c[8];
#pragma unroll
  for (int i = 0; i < 8; ++i) {
    v8f z0 = {0.f, 0.f, 0.f, 0.f, 0.f, 0.f, 0.f, 0.f};
    c[i] = z0;
  }

  // prologue: kick off async copy of W tile 0 into buffer 0
  {
    unsigned lds0 = bsBase + (unsigned)tid * 16u;
    async_load_b128(lds0,         (unsigned)tid * 16u,         Wp);
    async_load_b128(lds0 + 4096u, (unsigned)tid * 16u + 4096u, Wp);
  }

  for (int ks = 0; ks < KSTEPS; ++ks) {
    wait_asynccnt0();      // this thread's chunks of tile ks are in LDS
    __syncthreads();       // whole tile ks visible; prev readers all done

    if (ks + 1 < KSTEPS) { // overlap next tile's L2->LDS DMA with compute
      unsigned lds0 = bsBase + (unsigned)((ks + 1) & 1) * 8192u
                             + (unsigned)tid * 16u;
      const unsigned* gsrc = Wp + (size_t)(ks + 1) * 2048;
      async_load_b128(lds0,         (unsigned)tid * 16u,         gsrc);
      async_load_b128(lds0 + 4096u, (unsigned)tid * 16u + 4096u, gsrc);
    }

    // ---- build A fragment (16x32 bf16), contiguous K per lane half ----
    // lanes 0-15: K = {0..15}; lanes 16-31: K = {16..31}
    const int k0 = ks * 32;
    const float4* g0p = (const float4*)(xrow + k0 + hlf * 16);
    const float4* g1p = (const float4*)(xrow + k0 + 8 + hlf * 16);
    float4 f0 = g0p[0], f1 = g0p[1];
    float4 f2 = g1p[0], f3 = g1p[1];
    if (ks + 2 < KSTEPS)
      __builtin_prefetch(xrow + (ks + 2) * 32 + hlf * 8, 0, 0);

    v16bf a;
    a[0]  = (__bf16)f0.x; a[1]  = (__bf16)f0.y;
    a[2]  = (__bf16)f0.z; a[3]  = (__bf16)f0.w;
    a[4]  = (__bf16)f1.x; a[5]  = (__bf16)f1.y;
    a[6]  = (__bf16)f1.z; a[7]  = (__bf16)f1.w;
    a[8]  = (__bf16)f2.x; a[9]  = (__bf16)f2.y;
    a[10] = (__bf16)f2.z; a[11] = (__bf16)f2.w;
    a[12] = (__bf16)f3.x; a[13] = (__bf16)f3.y;
    a[14] = (__bf16)f3.z; a[15] = (__bf16)f3.w;

    // ---- 8 WMMAs: 16 rows x 128 cols, K += 32 ----
    const unsigned* BsCur = &Bs[ks & 1][0];
#pragma unroll
    for (int nt = 0; nt < 8; ++nt) {
      v16bf bfr = *(const v16bf*)(&BsCur[nt * 256 + lane * 8]);
      c[nt] = __builtin_amdgcn_wmma_f32_16x16x32_bf16(
          false, a, false, bfr, (short)0, c[nt], false, false);
    }
  }

  // ---- epilogue: write h + bias to LDS (C layout: row = v + 8*hlf) ----
#pragma unroll
  for (int nt = 0; nt < 8; ++nt) {
    int col = nt * 16 + l15;
    float bb = bias[col];
#pragma unroll
    for (int v = 0; v < 8; ++v) {
      int r = wave * 16 + hlf * 8 + v;
      hs[r][col] = c[nt][v] + bb;
    }
  }
  if (tid < TILE_M) {
    int rr = rowBase + tid;
    gs[tid] = (rr < N) ? batch[rr] : -1;
  }
  __syncthreads();

  // ---- LayerNorm + SiLU, 2 threads per row ----
  {
    int r = tid >> 1;
    int sub = tid & 1;
    float s = 0.f, q = 0.f;
#pragma unroll 8
    for (int i = 0; i < 64; ++i) {
      float v = hs[r][sub * 64 + i];
      s += v; q += v * v;
    }
    s += __shfl_xor(s, 1, 32);
    q += __shfl_xor(q, 1, 32);
    float mean = s * (1.f / 128.f);
    float var  = q * (1.f / 128.f) - mean * mean;
    float inv  = rsqrtf(var + LN_EPSF);
    for (int i = 0; i < 64; ++i) {
      int col = sub * 64 + i;
      float t = (hs[r][col] - mean) * inv * gamma[col] + beta[col];
      float zz = t / (1.f + __expf(-t));   // SiLU
      hs[r][col] = zz;
    }
  }
  __syncthreads();

  // ---- segmented pooling: run-length flush (batch is sorted) ----
  {
    int col = tid & 127;
    int r0  = (tid >> 7) * 64;          // two threads per column split rows
    int g   = gs[r0];
    if (g >= 0) {
      float rs = 0.f, rm = -3.402823466e38f;
      int rc = 0;
      for (int rr = r0; rr < r0 + 64; ++rr) {
        int gg = gs[rr];
        if (gg < 0) break;              // past N (sorted -> sentinel at end)
        if (gg != g) {
          atomicAdd(&z_sum[(size_t)g * P_DIM + col], rs);
          atomicMax(&z_maxu[(size_t)g * P_DIM + col], f2ord(rm));
          if (col == 0) atomicAdd(&z_cnt[g], (float)rc);
          g = gg; rs = 0.f; rm = -3.402823466e38f; rc = 0;
        }
        float zz = hs[rr][col];
        rs += zz; rm = fmaxf(rm, zz); ++rc;
      }
      if (rc > 0) {
        atomicAdd(&z_sum[(size_t)g * P_DIM + col], rs);
        atomicMax(&z_maxu[(size_t)g * P_DIM + col], f2ord(rm));
        if (col == 0) atomicAdd(&z_cnt[g], (float)rc);
      }
    }
  }
}

// ---- kernel 3: finalize mean/max, write [G, 2P] --------------------------

__global__ void esm_finalize_kernel(const float* __restrict__ z_sum,
                                    const float* __restrict__ z_cnt,
                                    const unsigned* __restrict__ z_maxu,
                                    float* __restrict__ out) {
  int i = blockIdx.x * 256 + threadIdx.x;   // 0 .. G*P-1
  int g = i >> 7, col = i & 127;
  float cnt  = z_cnt[g];
  float mean = (cnt > 0.f) ? z_sum[i] / cnt : 0.f;
  float mx   = (cnt > 0.f) ? ord2f(z_maxu[i]) : 0.f;
  out[(size_t)g * (2 * P_DIM) + col]          = mean;
  out[(size_t)g * (2 * P_DIM) + P_DIM + col]  = mx;
}

// ---- launcher ------------------------------------------------------------

extern "C" void kernel_launch(void* const* d_in, const int* in_sizes, int n_in,
                              void* d_out, int out_size, void* d_ws, size_t ws_size,
                              hipStream_t stream) {
  const float* x     = (const float*)d_in[0];
  const int*   batch = (const int*)d_in[1];   // jnp int64 -> int32 (x64 off)
  const float* W     = (const float*)d_in[2];
  const float* b     = (const float*)d_in[3];
  const float* gamma = (const float*)d_in[4];
  const float* beta  = (const float*)d_in[5];
  float* out = (float*)d_out;
  const int N = in_sizes[1];

  // workspace layout (2.44 MB total)
  char* ws = (char*)d_ws;
  unsigned* Wp     = (unsigned*)ws;                                 // 327680 B
  float*    z_sum  = (float*)(ws + 327680);                        // 1048576 B
  float*    z_cnt  = (float*)(ws + 327680 + 1048576);              //   8192 B
  unsigned* z_maxu = (unsigned*)(ws + 327680 + 1048576 + 8192);    // 1048576 B

  hipLaunchKernelGGL(esm_init_kernel, dim3((G_NUM * P_DIM + 255) / 256),
                     dim3(256), 0, stream, z_sum, z_cnt, z_maxu);
  hipLaunchKernelGGL(esm_packW_kernel, dim3(KSTEPS * 2048 / 256),
                     dim3(256), 0, stream, W, Wp);
  int nblocks = (N + TILE_M - 1) / TILE_M;
  hipLaunchKernelGGL(esm_gemm_ln_silu_pool_kernel, dim3(nblocks), dim3(256), 0,
                     stream, x, batch, Wp, b, gamma, beta,
                     z_sum, z_cnt, z_maxu, N);
  hipLaunchKernelGGL(esm_finalize_kernel, dim3(G_NUM * P_DIM / 256),
                     dim3(256), 0, stream, z_sum, z_cnt, z_maxu, out);
}
